// SynaptoGenesis_45878840656023
// MI455X (gfx1250) — matrix operand order repaired and checked by reading, not compiled, not run on hardware
//
#include <hip/hip_runtime.h>
#include <hip/hip_bf16.h>

#define N 8192
#define E 8192
#define D 128
#define NT (N / 16)          // 512 column tiles
#define TILE_PITCH 272       // 256B row + 16B TDM pad -> bank-conflict-free LDS reads

typedef __bf16 bf16_t;
typedef __attribute__((ext_vector_type(16))) __bf16 v16bf;
typedef __attribute__((ext_vector_type(8)))  __bf16 v8bf;
typedef __attribute__((ext_vector_type(8)))  float  v8f;

#if defined(__HIP_DEVICE_COMPILE__) && __has_builtin(__builtin_amdgcn_tensor_load_to_lds)
#define HAVE_TDM 1
#else
#define HAVE_TDM 0
#endif

// ---------- helpers ----------
__device__ __forceinline__ unsigned hashu(unsigned x) {
    x ^= x >> 17; x *= 0xed5ad4bbu;
    x ^= x >> 11; x *= 0xac4c1b51u;
    x ^= x >> 15; x *= 0x31848babu;
    x ^= x >> 14; return x;
}
__device__ __forceinline__ float u01(unsigned h) {
    return ((h >> 8) + 0.5f) * (1.0f / 16777216.0f);   // (0,1)
}

// A-matrix 16x32 bf16 fragment: lane half picks 16 contiguous K (matches B).
__device__ __forceinline__ v16bf load_a(const bf16_t* rowp, int kk, int hi) {
    v8bf lo = *(const v8bf*)(rowp + kk + 16 * hi);
    v8bf hv = *(const v8bf*)(rowp + kk + 16 * hi + 8);
    v16bf a;
#pragma unroll
    for (int i = 0; i < 8; ++i) { a[i] = lo[i]; a[i + 8] = hv[i]; }
    return a;
}
// B-matrix 32x16 bf16 fragment from global: 16 contiguous K at fixed column.
__device__ __forceinline__ v16bf load_b(const bf16_t* colp, int kk, int hi) {
    return *(const v16bf*)(colp + kk + 16 * hi);
}
// B fragment from the padded LDS tile (row l16 of tile = column l16).
__device__ __forceinline__ v16bf load_b_lds(const unsigned char* tb, int l16, int kk, int hi) {
    const unsigned char* p = tb + l16 * TILE_PITCH + (kk + 16 * hi) * 2;
    v8bf lo = *(const v8bf*)(p);
    v8bf hv = *(const v8bf*)(p + 16);
    v16bf b;
#pragma unroll
    for (int i = 0; i < 8; ++i) { b[i] = lo[i]; b[i + 8] = hv[i]; }
    return b;
}
__device__ __forceinline__ v8f wmma_bf16(v16bf a, v16bf b, v8f c) {
    return __builtin_amdgcn_wmma_f32_16x16x32_bf16(false, a, false, b, (short)0, c, false, false);
}

#if HAVE_TDM
typedef __attribute__((ext_vector_type(4))) unsigned tdm_u32x4;
typedef __attribute__((ext_vector_type(8))) int      tdm_i32x8;
typedef __attribute__((ext_vector_type(4))) int      tdm_i32x4;

// TDM D#: 2D tile, 16 rows x 128 bf16, row stride 128 elems in memory,
// LDS padding: 4 dwords after every 64 dwords -> 272B LDS row pitch.
__device__ __forceinline__ void tdm_load_tile(void* lds_dst, const void* gsrc) {
    unsigned lds_off = (unsigned)(unsigned long long)lds_dst;   // low 32 bits = LDS byte offset
    unsigned long long ga = (unsigned long long)gsrc;
    tdm_u32x4 g0;
    g0[0] = 1u;                                    // count=1 (valid user descriptor)
    g0[1] = lds_off;                               // lds_addr
    g0[2] = (unsigned)ga;                          // global_addr[31:0]
    g0[3] = (unsigned)(ga >> 32) | (2u << 30);     // global_addr[56:32] | type=2 (image)
    tdm_i32x8 g1;
    g1[0] = (1 << 16)                              // data_size = 2 bytes
          | (1 << 20)                              // pad_enable
          | (5 << 22)                              // pad_interval: every 64 dwords (256B)
          | (3 << 25);                             // pad_amount: 4 dwords (16B)
    g1[1] = (int)((128u & 0xFFFFu) << 16);         // tensor_dim0[15:0] = 128
    g1[2] = (int)((8192u & 0xFFFFu) << 16);        // tensor_dim1[15:0] = 8192
    g1[3] = (int)(128u << 16);                     // tile_dim0 = 128
    g1[4] = 16;                                    // tile_dim1 = 16
    g1[5] = 128;                                   // tensor_dim0_stride = 128
    g1[6] = 0;
    g1[7] = 0;
    tdm_i32x4 z4 = {};
#if __clang_major__ >= 23
    tdm_i32x8 z8 = {};
    __builtin_amdgcn_tensor_load_to_lds(g0, g1, z4, z4, z8, 0);
#else
    __builtin_amdgcn_tensor_load_to_lds(g0, g1, z4, z4, 0);
#endif
}
#endif

// ---------- kernels ----------
__global__ void k_conv_nodes(const float* __restrict__ nodes, bf16_t* __restrict__ nb) {
    int i = blockIdx.x * blockDim.x + threadIdx.x;
    if (i < N * D) nb[i] = (bf16_t)nodes[i];
}

__global__ void k_conv_wqt(const float* __restrict__ Wq, bf16_t* __restrict__ wqt) {
    int i = blockIdx.x * blockDim.x + threadIdx.x;     // i = n*128 + k
    if (i < D * D) {
        int n = i >> 7, k = i & 127;
        wqt[i] = (bf16_t)Wq[k * D + n];                // WqT[n][k] = Wq[k][n]
    }
}

// one wave per node: prob = sigmoid(h . Wp + bp); gens = (u < prob * anodes)
__global__ void k_probs_gens(const float* __restrict__ nodes, const float* __restrict__ anodes,
                             const float* __restrict__ Wp, const float* __restrict__ bp,
                             int* __restrict__ gens) {
    int wid  = (blockIdx.x * blockDim.x + threadIdx.x) >> 5;
    int lane = threadIdx.x & 31;
    if (wid >= N) return;
    float4 x = *(const float4*)(nodes + wid * D + lane * 4);
    float4 w = *(const float4*)(Wp + lane * 4);
    float s = x.x * w.x + x.y * w.y + x.z * w.z + x.w * w.w;
#pragma unroll
    for (int m = 16; m >= 1; m >>= 1) s += __shfl_xor(s, m, 32);
    if (lane == 0) {
        float p = 1.0f / (1.0f + __expf(-(s + bp[0])));
        float u = u01(hashu(0x9e3779b9u ^ (unsigned)wid));
        gens[wid] = (u < p * anodes[wid]) ? 1 : 0;
    }
}

// single block: e_active, exclusive scan of gens, n_gens, scatter targets
__global__ void k_scan(const int* __restrict__ gens, const float* __restrict__ aedges,
                       int* __restrict__ trg, int* __restrict__ scal) {
    __shared__ int   part[1024];
    __shared__ float fpart[1024];
    int t = threadIdx.x;
    float fs = 0.f;
#pragma unroll
    for (int i = 0; i < 8; ++i) fs += aedges[t * 8 + i];
    fpart[t] = fs; __syncthreads();
    for (int off = 512; off > 0; off >>= 1) { if (t < off) fpart[t] += fpart[t + off]; __syncthreads(); }
    int e_active = (int)(fpart[0] + 0.5f);
    __syncthreads();

    int loc[8], s = 0;
#pragma unroll
    for (int i = 0; i < 8; ++i) { loc[i] = s; s += gens[t * 8 + i]; }
    part[t] = s; __syncthreads();
    for (int off = 1; off < 1024; off <<= 1) {            // inclusive Hillis-Steele
        int v = part[t];
        int a = (t >= off) ? part[t - off] : 0;
        __syncthreads();
        part[t] = v + a; __syncthreads();
    }
    int total = part[1023];
    int prev  = (t > 0) ? part[t - 1] : 0;
#pragma unroll
    for (int i = 0; i < 8; ++i) {
        int idx = t * 8 + i;
        trg[idx] = gens[idx] ? (e_active + prev + loc[i]) : -1;
    }
    if (t == 0) {
        int allowed = E - e_active - 1;
        int ng = total < 0 ? 0 : (total > allowed ? allowed : total);
        scal[0] = e_active; scal[1] = ng;
    }
}

// queries = nodes @ Wq + bq, via WMMA bf16 -> bf16 output. One wave per 16x16 tile.
__global__ void k_queries(const bf16_t* __restrict__ nb, const bf16_t* __restrict__ wqt,
                          const float* __restrict__ bq, bf16_t* __restrict__ qb) {
    int wid  = (blockIdx.x * blockDim.x + threadIdx.x) >> 5;
    int lane = threadIdx.x & 31;
    int mt = wid >> 3, nt = wid & 7;
    int hi = lane >> 4, l16 = lane & 15;
    const bf16_t* arow = nb  + (mt * 16 + l16) * D;
    const bf16_t* brow = wqt + (nt * 16 + l16) * D;
    v8f acc = {};
#pragma unroll
    for (int kki = 0; kki < 4; ++kki) {
        int kk = kki * 32;
        acc = wmma_bf16(load_a(arow, kk, hi), load_b(brow, kk, hi), acc);
    }
    float bias = bq[nt * 16 + l16];
#pragma unroll
    for (int r = 0; r < 8; ++r) {
        int orow = mt * 16 + r + 8 * hi;                 // C layout: VGPR r -> row r (+8 upper)
        qb[orow * D + (nt * 16 + l16)] = (bf16_t)(acc[r] + bias);
    }
}

// scores = queries @ nodes^T fused with clip, active-mask penalty, Gumbel-max sampling.
// Block = 8 waves sharing TDM-staged, double-buffered B tiles in LDS; each wave owns a
// 16-row stripe with its A fragments register-resident across all 512 N-tiles.
__global__ void k_scores_select(const bf16_t* __restrict__ qb, const bf16_t* __restrict__ nb,
                                const float* __restrict__ anodes, const int* __restrict__ gens,
                                int* __restrict__ select) {
    __shared__ __align__(16) unsigned char smem[2][16 * TILE_PITCH];
    int tid  = threadIdx.x;
    int wave = tid >> 5, lane = tid & 31;
    int hi = lane >> 4, l16 = lane & 15;
    int mrow0 = blockIdx.x * 128 + wave * 16;

    const bf16_t* arow = qb + (mrow0 + l16) * D;
    v16bf a0 = load_a(arow, 0, hi), a1 = load_a(arow, 32, hi);
    v16bf a2 = load_a(arow, 64, hi), a3 = load_a(arow, 96, hi);

    float best[8]; int barg[8];
#pragma unroll
    for (int r = 0; r < 8; ++r) { best[r] = -3.4e38f; barg[r] = 0; }

#if HAVE_TDM
    bool issuer = (wave == 0);
    if (issuer) {
        tdm_load_tile(smem[0], nb);                      // prime tile 0
        __builtin_amdgcn_s_wait_tensorcnt(0);
    }
    __syncthreads();
#endif

    for (int nt = 0; nt < NT; ++nt) {
#if HAVE_TDM
        int cur = nt & 1;
        if (issuer && nt + 1 < NT)                       // overlap DMA of next tile
            tdm_load_tile(smem[cur ^ 1], nb + (nt + 1) * 16 * D);
        const unsigned char* tb = smem[cur];
#else
        // cooperative synchronous staging: 256 threads x 16B = one 16x256B tile
        __syncthreads();
        {
            int row = tid >> 4, chunk = tid & 15;
            *(uint4*)(smem[0] + row * TILE_PITCH + chunk * 16) =
                *(const uint4*)(nb + nt * 16 * D + row * D + chunk * 8);
        }
        __syncthreads();
        const unsigned char* tb = smem[0];
#endif
        int col = nt * 16 + l16;
        v8f acc = {};
        acc = wmma_bf16(a0, load_b_lds(tb, l16,  0, hi), acc);
        acc = wmma_bf16(a1, load_b_lds(tb, l16, 32, hi), acc);
        acc = wmma_bf16(a2, load_b_lds(tb, l16, 64, hi), acc);
        acc = wmma_bf16(a3, load_b_lds(tb, l16, 96, hi), acc);
        float pen = (1.0f - anodes[col]) * 1e10f;
#pragma unroll
        for (int r = 0; r < 8; ++r) {
            int row = mrow0 + r + 8 * hi;
            float s = fminf(fmaxf(acc[r], -1e4f), 1e4f) - pen;
            unsigned h = hashu(0x27d4eb2fu ^ (unsigned)(row * N + col));
            s += -__logf(-__logf(u01(h)));               // Gumbel-max == categorical
            if (s > best[r]) { best[r] = s; barg[r] = col; }
        }
#if HAVE_TDM
        if (issuer && nt + 1 < NT)
            __builtin_amdgcn_s_wait_tensorcnt(0);        // next tile landed
        __syncthreads();                                  // hand buffers over
#endif
    }

#pragma unroll
    for (int r = 0; r < 8; ++r) {                        // reduce across 16 column-lanes
        float bv = best[r]; int ba = barg[r];
#pragma unroll
        for (int m = 1; m < 16; m <<= 1) {
            float ov = __shfl_xor(bv, m, 32);
            int   oa = __shfl_xor(ba, m, 32);
            if (ov > bv || (ov == bv && oa < ba)) { bv = ov; ba = oa; }
        }
        if (l16 == 0) {
            int row = mrow0 + r + 8 * hi;
            select[row] = gens[row] ? ba : 0;
        }
    }
}

// new_edges noise, naedges, base nsend/nrec. One block (128 threads) per edge row.
__global__ void k_edges(const float* __restrict__ edges, const int* __restrict__ send,
                        const int* __restrict__ rec, const int* __restrict__ scal,
                        float* __restrict__ out_edges, int* __restrict__ out_send,
                        int* __restrict__ out_rec, float* __restrict__ out_ae) {
    int e = blockIdx.x, d = threadIdx.x;
    int e_active = scal[0], ng = scal[1];
    bool act  = (e < e_active + ng) && (e != E - 1);
    bool mnew = act && (e >= e_active);
    float noise = 0.f;
    if (mnew) {
        unsigned idx = (unsigned)(e * D + d);
        float u1 = u01(hashu(0x85ebca6bu ^ idx));
        float u2 = u01(hashu(0xc2b2ae35u ^ idx));
        noise = __fsqrt_rn(-2.0f * __logf(u1)) * __cosf(6.28318530718f * u2);
    }
    out_edges[e * D + d] = edges[e * D + d] + noise;
    if (d == 0) {
        out_ae[e]   = act ? 1.0f : 0.0f;
        out_send[e] = mnew ? 0 : send[e];
        out_rec[e]  = mnew ? 0 : rec[e];
    }
}

__global__ void k_scatter(const int* __restrict__ gens, const int* __restrict__ trg,
                          const int* __restrict__ select, int* __restrict__ out_send,
                          int* __restrict__ out_rec) {
    int i = blockIdx.x * blockDim.x + threadIdx.x;
    if (i >= N || !gens[i]) return;
    int t = trg[i];
    if (t < 0 || t >= E) return;
    atomicAdd(&out_send[t], i);
    atomicAdd(&out_rec[t], select[i]);
}

// ---------- launch ----------
extern "C" void kernel_launch(void* const* d_in, const int* in_sizes, int n_in,
                              void* d_out, int out_size, void* d_ws, size_t ws_size,
                              hipStream_t stream) {
    const float* nodes  = (const float*)d_in[0];
    const float* edges  = (const float*)d_in[1];
    const int*   rec    = (const int*)  d_in[2];
    const int*   send   = (const int*)  d_in[3];
    const float* anodes = (const float*)d_in[4];
    const float* aedges = (const float*)d_in[5];
    const float* Wq     = (const float*)d_in[6];
    const float* bq     = (const float*)d_in[7];
    const float* Wp     = (const float*)d_in[8];
    const float* bp     = (const float*)d_in[9];

    char* w = (char*)d_ws;
    bf16_t* nb   = (bf16_t*)(w + 0);                 // 2 MB
    bf16_t* qb   = (bf16_t*)(w + 2097152);           // 2 MB
    bf16_t* wqt  = (bf16_t*)(w + 4194304);           // 32 KB
    int*    gens = (int*)   (w + 4227072);           // 32 KB
    int*    trg  = (int*)   (w + 4259840);           // 32 KB
    int*    sel  = (int*)   (w + 4292608);           // 32 KB
    int*    scal = (int*)   (w + 4325376);           // scalars

    float* out_edges = (float*)d_out;                // [E*D] f32
    int*   out_send  = (int*)((float*)d_out + E * D);// [E] i32
    int*   out_rec   = out_send + E;                 // [E] i32
    float* out_ae    = (float*)(out_rec + E);        // [E] f32

    k_conv_nodes <<<(N * D + 255) / 256, 256, 0, stream>>>(nodes, nb);
    k_conv_wqt   <<<(D * D + 255) / 256, 256, 0, stream>>>(Wq, wqt);
    k_probs_gens <<<N * 32 / 256, 256, 0, stream>>>(nodes, anodes, Wp, bp, gens);
    k_scan       <<<1, 1024, 0, stream>>>(gens, aedges, trg, scal);
    k_queries    <<<(N / 16) * (D / 16) * 32 / 256, 256, 0, stream>>>(nb, wqt, bq, qb);
    k_scores_select<<<N / 128, 256, 0, stream>>>(qb, nb, anodes, gens, sel);
    k_edges      <<<E, D, 0, stream>>>(edges, send, rec, scal, out_edges, out_send, out_rec, out_ae);
    k_scatter    <<<N / 256, 256, 0, stream>>>(gens, trg, sel, out_send, out_rec);
}
